// MultiHeadAttention_11785390260651
// MI455X (gfx1250) — matrix operand order repaired and checked by reading, not compiled, not run on hardware
//
#include <hip/hip_runtime.h>
#include <hip/hip_bf16.h>

#define DEV __device__ __forceinline__

typedef __attribute__((ext_vector_type(16))) _Float16 v16h;
typedef __attribute__((ext_vector_type(8)))  _Float16 v8h;
typedef __attribute__((ext_vector_type(8)))  float    v8f;
typedef __attribute__((ext_vector_type(4)))  float    v4f;
typedef __attribute__((ext_vector_type(4)))  unsigned u32x4;
typedef __attribute__((ext_vector_type(8)))  int      i32x8;
typedef __attribute__((ext_vector_type(4)))  int      i32x4;

union V16u { v16h v; v8h h[2]; };

static constexpr int Bdim = 8, Sdim = 1024, Ddim = 1024, Hh = 16, DHd = 64;
static constexpr int BS = Bdim * Sdim; // 8192

#if __has_builtin(__builtin_amdgcn_tensor_load_to_lds)
#define HAVE_TDM 1
#else
#define HAVE_TDM 0
#endif

// ---- WMMA fragment helpers (wave32, 16x16x32 f16 shapes per CDNA5 ISA 7.12.2) ----
// A (16x32, MxK) per-lane: l=lane&15 -> row M=l; g=lane>>4 -> K run [16g .. 16g+15] contiguous
DEV v16h load_a_frag(const _Float16* base, int rowStride) {
  const int lane = threadIdx.x & 31;
  const int l = lane & 15, g = lane >> 4;
  const _Float16* p = base + (size_t)l * rowStride;
  V16u r;
  r.v = *(const v16h*)(p + 16 * g);
  return r.v;
}

// B (32x16, KxN) per-lane: col N = lane&15; K run [16g .. 16g+15] contiguous.
DEV v16h load_b_frag(const _Float16* base, int rowStride) {
  const int lane = threadIdx.x & 31;
  const int l = lane & 15, g = lane >> 4;
  return *(const v16h*)(base + (size_t)l * rowStride + 16 * g);
}

DEV v8f wmma_f16(v16h a, v16h b, v8f c) {
  return __builtin_amdgcn_wmma_f32_16x16x32_f16(false, a, false, b, (short)0, c, false, false);
}

// pack 8 f32 -> v8h (single 16B LDS store; converts can fuse to packed cvt)
DEV v8h cvt8(v4f a, v4f b) {
  v8h h;
  h[0] = (_Float16)a.x; h[1] = (_Float16)a.y; h[2] = (_Float16)a.z; h[3] = (_Float16)a.w;
  h[4] = (_Float16)b.x; h[5] = (_Float16)b.y; h[6] = (_Float16)b.z; h[7] = (_Float16)b.w;
  return h;
}

#if HAVE_TDM
// Tensor Data Mover: 2D f16 tile (tileW x tileH elems), row stride in elems.
// D# layout per CDNA5 ISA.
DEV void tdm_load_2d_f16(unsigned lds_off, const void* gaddr, int strideElems,
                         int tileW, int tileH, int tensorW, int tensorH) {
  const unsigned long long ga = (unsigned long long)(size_t)gaddr;
  u32x4 g0;
  g0[0] = 1u;                                                   // count=1, user desc
  g0[1] = lds_off;                                              // lds_addr
  g0[2] = (unsigned)ga;                                         // global_addr[31:0]
  g0[3] = (unsigned)((ga >> 32) & 0x01FFFFFFu) | (2u << 30);    // global_addr[56:32] | type=2
  i32x8 g1;
  g1[0] = 1 << 16;                                              // data_size=1 (2B), mask=0
  g1[1] = (int)(((unsigned)tensorW & 0xFFFFu) << 16);           // tensor_dim0[15:0]
  g1[2] = (int)((((unsigned)tensorW >> 16) & 0xFFFFu) |
                (((unsigned)tensorH & 0xFFFFu) << 16));         // dim0 hi | dim1 lo
  g1[3] = (int)((((unsigned)tensorH >> 16) & 0xFFFFu) |
                (((unsigned)tileW & 0xFFFFu) << 16));           // dim1 hi | tile_dim0
  g1[4] = tileH & 0xFFFF;                                       // tile_dim1 (tile_dim2=0)
  g1[5] = strideElems;                                          // tensor_dim0_stride lo
  g1[6] = 0;                                                    // stride0 hi | stride1 lo
  g1[7] = 0;
  i32x4 z4 = {0, 0, 0, 0};
#if defined(__clang_major__) && (__clang_major__ >= 23)
  i32x8 z8 = {0, 0, 0, 0, 0, 0, 0, 0};
  __builtin_amdgcn_tensor_load_to_lds(g0, g1, z4, z4, z8, 0);
#else
  __builtin_amdgcn_tensor_load_to_lds(g0, g1, z4, z4, 0);
#endif
}

DEV void tdm_wait() {
#if __has_builtin(__builtin_amdgcn_s_wait_tensorcnt)
  __builtin_amdgcn_s_wait_tensorcnt(0);
#else
  asm volatile("s_wait_tensorcnt 0x0" ::: "memory");
#endif
}
#endif  // HAVE_TDM

// =============================================================================
// Kernel 1: projection GEMM  Out = op(X @ W^T + b)   (f32 in, f16 out)
// Block tile 128x128, 8 waves of 64x32, BLOCK_K=32. OP: 0=identity, 1=log_sigmoid
// =============================================================================
template <int OP>
__global__ __launch_bounds__(256) void proj_gemm(const float* __restrict__ X,
                                                 const float* __restrict__ W,
                                                 const float* __restrict__ bias,
                                                 _Float16* __restrict__ Out) {
  __shared__ __align__(32) _Float16 sX[128 * 32];
  __shared__ __align__(32) _Float16 sW[128 * 32];

  const int blockN = blockIdx.x * 128;
  const int blockM = blockIdx.y * 128;
  const int t = threadIdx.x;
  const int wave = t >> 5;
  const int waveM = (wave >> 2) * 64;
  const int waveN = (wave & 3) * 32;

  const int lrow = t >> 1;        // 0..127
  const int lcol = (t & 1) * 16;  // 0 or 16

  v8f acc[4][2] = {};

  for (int k0 = 0; k0 < Ddim; k0 += 32) {
    {
      const float* srcX = X + (size_t)(blockM + lrow) * Ddim + k0 + lcol;
      const float* srcW = W + (size_t)(blockN + lrow) * Ddim + k0 + lcol;
      _Float16* dstX = sX + lrow * 32 + lcol;
      _Float16* dstW = sW + lrow * 32 + lcol;
      *(v8h*)(dstX + 0) = cvt8(*(const v4f*)(srcX + 0), *(const v4f*)(srcX + 4));
      *(v8h*)(dstX + 8) = cvt8(*(const v4f*)(srcX + 8), *(const v4f*)(srcX + 12));
      *(v8h*)(dstW + 0) = cvt8(*(const v4f*)(srcW + 0), *(const v4f*)(srcW + 4));
      *(v8h*)(dstW + 8) = cvt8(*(const v4f*)(srcW + 8), *(const v4f*)(srcW + 12));
      if (k0 + 32 < Ddim) {  // global_prefetch_b8 of next K chunk
        __builtin_prefetch(srcX + 32, 0, 0);
        __builtin_prefetch(srcW + 32, 0, 0);
      }
    }
    __syncthreads();

    v16h bfr[2];
#pragma unroll
    for (int ni = 0; ni < 2; ++ni)
      bfr[ni] = load_b_frag(sW + (waveN + 16 * ni) * 32, 32);
#pragma unroll
    for (int mi = 0; mi < 4; ++mi) {
      v16h a = load_a_frag(sX + (waveM + 16 * mi) * 32, 32);
#pragma unroll
      for (int ni = 0; ni < 2; ++ni)
        acc[mi][ni] = wmma_f16(a, bfr[ni], acc[mi][ni]);
    }
    __syncthreads();
  }

  // epilogue: C layout (lane g,l): VGPR r -> row M = 8g + r, col N = l
  const int lane = t & 31, l = lane & 15, g = lane >> 4;
#pragma unroll
  for (int mi = 0; mi < 4; ++mi) {
#pragma unroll
    for (int ni = 0; ni < 2; ++ni) {
      const int col = blockN + waveN + 16 * ni + l;
      const float bv = bias[col];
#pragma unroll
      for (int r = 0; r < 8; ++r) {
        const int row = blockM + waveM + 16 * mi + 8 * g + r;
        float v = acc[mi][ni][r] + bv;
        if (OP == 1) v = fminf(v, 0.f) - log1pf(__expf(-fabsf(v)));
        Out[(size_t)row * Ddim + col] = (_Float16)v;
      }
    }
  }
}

// =============================================================================
// Kernel 2: flash attention per (b,h), 64 query rows / block, 4 waves,
// each wave owns 16 query rows. y = Rg * exp(softmax(QK^T/8) @ LSV)
// =============================================================================
__global__ __launch_bounds__(128) void attn_flash(const _Float16* __restrict__ Qh,
                                                  const _Float16* __restrict__ Kh,
                                                  const _Float16* __restrict__ LSV,
                                                  const _Float16* __restrict__ Rg,
                                                  _Float16* __restrict__ G) {
  __shared__ __align__(32) _Float16 sK[64 * 64];      // [t][dh]
  __shared__ __align__(32) _Float16 sVT[64 * 64];     // [dh][t] (transposed)
  __shared__ __align__(32) _Float16 sP[4][16 * 64];   // per-wave P tile [m][t]

  const int bh = blockIdx.y;          // 0..127
  const int b = bh / Hh, h = bh % Hh;
  const int qBase = blockIdx.x * 64;  // query row tile within S
  const int t = threadIdx.x;
  const int wave = t >> 5;
  const int lane = t & 31, l = lane & 15, g = lane >> 4;

  const size_t rowBase = (size_t)b * Sdim;
  const int colH = h * DHd;

  // Q fragments for this wave's 16 rows; fold 1/sqrt(DH)=0.125 (exact pow2) in once
  const _Float16* qptr = Qh + (rowBase + qBase + wave * 16) * Ddim + colH;
  v16h qa[2];
#pragma unroll
  for (int kk = 0; kk < 2; ++kk) {
    qa[kk] = load_a_frag(qptr + 32 * kk, Ddim);
#pragma unroll
    for (int i = 0; i < 16; ++i) qa[kk][i] *= (_Float16)0.125f;
  }

  v8f yacc[4] = {};
  float rowM[8], rowL[8];
#pragma unroll
  for (int r = 0; r < 8; ++r) { rowM[r] = -1e30f; rowL[r] = 0.f; }

  for (int t0 = 0; t0 < Sdim; t0 += 64) {
    // --- K tile: TDM async DMA (wave 0) overlapped with LSV transpose staging ---
#if HAVE_TDM
    if (wave == 0)
      tdm_load_2d_f16((unsigned)(size_t)(void*)sK,
                      Kh + (rowBase + t0) * Ddim + colH,
                      Ddim, /*tileW=*/64, /*tileH=*/64, /*tensorW=*/Ddim,
                      /*tensorH=*/Sdim);
#else
    {
      const int rr = t >> 1, cc = (t & 1) * 32;
      const _Float16* ksrc = Kh + (rowBase + t0 + rr) * Ddim + colH + cc;
      *(v8h*)(sK + rr * 64 + cc + 0)  = *(const v8h*)(ksrc + 0);
      *(v8h*)(sK + rr * 64 + cc + 8)  = *(const v8h*)(ksrc + 8);
      *(v8h*)(sK + rr * 64 + cc + 16) = *(const v8h*)(ksrc + 16);
      *(v8h*)(sK + rr * 64 + cc + 24) = *(const v8h*)(ksrc + 24);
    }
#endif
    // LSV tile staged transposed: vectorized 16B global loads, register scatter
    {
      const int rr = t >> 1, cc = (t & 1) * 32;
      const _Float16* vsrc = LSV + (rowBase + t0 + rr) * Ddim + colH + cc;
#pragma unroll
      for (int c = 0; c < 4; ++c) {
        v8h vv = *(const v8h*)(vsrc + 8 * c);
#pragma unroll
        for (int j = 0; j < 8; ++j) sVT[(cc + 8 * c + j) * 64 + rr] = vv[j];
      }
    }
#if HAVE_TDM
    if (wave == 0) tdm_wait();
#endif
    __syncthreads();

    // S = (Q/8) @ K^T : 4 n-frags of 16 keys, 2 k-steps over DH=64
    v8f s[4] = {};
#pragma unroll
    for (int kk = 0; kk < 2; ++kk) {
#pragma unroll
      for (int ni = 0; ni < 4; ++ni) {
        v16h bf = load_b_frag(sK + (16 * ni) * 64 + 32 * kk, 64);
        s[ni] = wmma_f16(qa[kk], bf, s[ni]);
      }
    }

    // online softmax: row stats per (g, r) -> row M = 8g + r
    float newM[8], alpha[8];
#pragma unroll
    for (int r = 0; r < 8; ++r) {
      float m = fmaxf(fmaxf(s[0][r], s[1][r]), fmaxf(s[2][r], s[3][r]));
#pragma unroll
      for (int off = 8; off >= 1; off >>= 1) m = fmaxf(m, __shfl_xor(m, off, 32));
      newM[r] = fmaxf(rowM[r], m);
      alpha[r] = __expf(rowM[r] - newM[r]);
    }

    _Float16* pbuf = sP[wave];
    float partial[8];
#pragma unroll
    for (int r = 0; r < 8; ++r) partial[r] = 0.f;
#pragma unroll
    for (int ni = 0; ni < 4; ++ni) {
#pragma unroll
      for (int r = 0; r < 8; ++r) {
        float p = __expf(s[ni][r] - newM[r]);
        partial[r] += p;
        pbuf[(8 * g + r) * 64 + 16 * ni + l] = (_Float16)p;  // C-layout -> [m][t]
      }
    }
#pragma unroll
    for (int r = 0; r < 8; ++r) {
      float ps = partial[r];
#pragma unroll
      for (int off = 8; off >= 1; off >>= 1) ps += __shfl_xor(ps, off, 32);
      rowL[r] = rowL[r] * alpha[r] + ps;
      rowM[r] = newM[r];
    }
#pragma unroll
    for (int ni = 0; ni < 4; ++ni)
#pragma unroll
      for (int r = 0; r < 8; ++r) yacc[ni][r] *= alpha[r];

    // Y += P @ LSV (A-frags restaged through LDS; B-frags from transposed tile)
#pragma unroll
    for (int kk = 0; kk < 2; ++kk) {
      v16h pa = load_a_frag(pbuf + 32 * kk, 64);
#pragma unroll
      for (int ni = 0; ni < 4; ++ni) {
        v16h bf = load_b_frag(sVT + (16 * ni) * 64 + 32 * kk, 64);
        yacc[ni] = wmma_f16(pa, bf, yacc[ni]);
      }
    }
    __syncthreads();
  }

  // epilogue: G = Rg * exp(Yacc / L)
#pragma unroll
  for (int ni = 0; ni < 4; ++ni) {
#pragma unroll
    for (int r = 0; r < 8; ++r) {
      const int row = qBase + wave * 16 + 8 * g + r;
      const int col = colH + 16 * ni + l;
      const size_t idx = (rowBase + row) * Ddim + col;
      const float y = __expf(yacc[ni][r] / rowL[r]);
      G[idx] = (_Float16)((float)Rg[idx] * y);
    }
  }
}

// =============================================================================
// Kernel 3: output GEMM  Out = G @ Wo^T + bo + residual   (f16 A, f32 out)
// =============================================================================
__global__ __launch_bounds__(256) void out_gemm(const _Float16* __restrict__ Gm,
                                                const float* __restrict__ W,
                                                const float* __restrict__ bias,
                                                const float* __restrict__ resid,
                                                float* __restrict__ Out) {
  __shared__ __align__(32) _Float16 sX[128 * 32];
  __shared__ __align__(32) _Float16 sW[128 * 32];

  const int blockN = blockIdx.x * 128;
  const int blockM = blockIdx.y * 128;
  const int t = threadIdx.x;
  const int wave = t >> 5;
  const int waveM = (wave >> 2) * 64;
  const int waveN = (wave & 3) * 32;

  const int lrow = t >> 1;
  const int lcol = (t & 1) * 16;

  v8f acc[4][2] = {};

  for (int k0 = 0; k0 < Ddim; k0 += 32) {
    {
      const _Float16* srcX = Gm + (size_t)(blockM + lrow) * Ddim + k0 + lcol;
      _Float16* dstX = sX + lrow * 32 + lcol;
      *(v8h*)(dstX + 0) = *(const v8h*)(srcX + 0);
      *(v8h*)(dstX + 8) = *(const v8h*)(srcX + 8);
      const float* srcW = W + (size_t)(blockN + lrow) * Ddim + k0 + lcol;
      _Float16* dstW = sW + lrow * 32 + lcol;
      *(v8h*)(dstW + 0) = cvt8(*(const v4f*)(srcW + 0), *(const v4f*)(srcW + 4));
      *(v8h*)(dstW + 8) = cvt8(*(const v4f*)(srcW + 8), *(const v4f*)(srcW + 12));
      if (k0 + 32 < Ddim) {
        __builtin_prefetch(srcX + 32, 0, 0);
        __builtin_prefetch(srcW + 32, 0, 0);
      }
    }
    __syncthreads();

    v16h bfr[2];
#pragma unroll
    for (int ni = 0; ni < 2; ++ni)
      bfr[ni] = load_b_frag(sW + (waveN + 16 * ni) * 32, 32);
#pragma unroll
    for (int mi = 0; mi < 4; ++mi) {
      v16h a = load_a_frag(sX + (waveM + 16 * mi) * 32, 32);
#pragma unroll
      for (int ni = 0; ni < 2; ++ni)
        acc[mi][ni] = wmma_f16(a, bfr[ni], acc[mi][ni]);
    }
    __syncthreads();
  }

  const int lane = t & 31, l = lane & 15, g = lane >> 4;
#pragma unroll
  for (int mi = 0; mi < 4; ++mi) {
#pragma unroll
    for (int ni = 0; ni < 2; ++ni) {
      const int col = blockN + waveN + 16 * ni + l;
      const float bv = bias[col];
#pragma unroll
      for (int r = 0; r < 8; ++r) {
        const int row = blockM + waveM + 16 * mi + 8 * g + r;
        const size_t idx = (size_t)row * Ddim + col;
        Out[idx] = acc[mi][ni][r] + bv + resid[idx];
      }
    }
  }
}

// =============================================================================
extern "C" void kernel_launch(void* const* d_in, const int* in_sizes, int n_in,
                              void* d_out, int out_size, void* d_ws, size_t ws_size,
                              hipStream_t stream) {
  const float* X  = (const float*)d_in[0];
  const float* Wq = (const float*)d_in[1];  const float* bq = (const float*)d_in[2];
  const float* Wk = (const float*)d_in[3];  const float* bk = (const float*)d_in[4];
  const float* Wv = (const float*)d_in[5];  const float* bv = (const float*)d_in[6];
  const float* Wr = (const float*)d_in[7];  const float* br = (const float*)d_in[8];
  const float* Wo = (const float*)d_in[9];  const float* bo = (const float*)d_in[10];
  float* Out = (float*)d_out;

  // scratch: 5 f16 planes of [BS, D] = 5 * 16MB = 80MB
  char* ws = (char*)d_ws;
  const size_t planeH = (size_t)BS * Ddim * sizeof(_Float16);
  _Float16* Qh  = (_Float16*)(ws + 0 * planeH);
  _Float16* Kh  = (_Float16*)(ws + 1 * planeH);
  _Float16* LSV = (_Float16*)(ws + 2 * planeH);  // log_sigmoid(V)
  _Float16* Rg  = (_Float16*)(ws + 3 * planeH);
  _Float16* G   = (_Float16*)(ws + 4 * planeH);  // gated attention output

  dim3 gblk(Ddim / 128, BS / 128);  // (8, 64)
  proj_gemm<0><<<gblk, 256, 0, stream>>>(X, Wq, bq, Qh);
  proj_gemm<0><<<gblk, 256, 0, stream>>>(X, Wk, bk, Kh);
  proj_gemm<1><<<gblk, 256, 0, stream>>>(X, Wv, bv, LSV);
  proj_gemm<0><<<gblk, 256, 0, stream>>>(X, Wr, br, Rg);

  dim3 ablk(Sdim / 64, Bdim * Hh);  // (16, 128)
  attn_flash<<<ablk, 128, 0, stream>>>(Qh, Kh, LSV, Rg, G);

  out_gemm<<<gblk, 256, 0, stream>>>(G, Wo, bo, X, Out);
}
